// GCN_layers_28226525069657
// MI455X (gfx1250) — matrix operand order repaired and checked by reading, not compiled, not run on hardware
//
#include <hip/hip_runtime.h>
#include <hip/hip_bf16.h>
#include <math.h>

typedef __attribute__((ext_vector_type(16))) __bf16 v16bf;
typedef __attribute__((ext_vector_type(8)))  __bf16 v8bf;
typedef __attribute__((ext_vector_type(8)))  float  v8f;

#define N_NODES 50000
#define N_EDGES 800000
#define IN_DIM  768
#define HID     32
#define HEADS   4
#define OUT_DIM 128
// both layers have H*D == 128 feature columns
#define FEAT    128

// ---------------------------------------------------------------------------
// Elementwise helpers
// ---------------------------------------------------------------------------
__global__ void k_f32_to_bf16(const float* __restrict__ in, __bf16* __restrict__ out, int n) {
    int i = blockIdx.x * blockDim.x + threadIdx.x;
    if (i < n) out[i] = (__bf16)in[i];
}

__global__ void k_fill(float* __restrict__ p, float v, int n) {
    int i = blockIdx.x * blockDim.x + threadIdx.x;
    if (i < n) p[i] = v;
}

// ELU on layer-1 accumulator, emit bf16 for the layer-2 GEMM
__global__ void k_elu_to_bf16(const float* __restrict__ in, __bf16* __restrict__ out, int n) {
    int i = blockIdx.x * blockDim.x + threadIdx.x;
    if (i < n) {
        float x = in[i];
        float y = (x > 0.0f) ? x : expm1f(x);
        out[i] = (__bf16)y;
    }
}

// ---------------------------------------------------------------------------
// Pack weights into the per-lane WMMA B-fragment layout:
//   Bp[((nt*KT + kt)*32 + lane)*16 + e]  holds  B[k][n]  with
//   n = nt*16 + (lane&15),  k = kt*32 + (lane>>4)*16 + e
// ---------------------------------------------------------------------------
__global__ void k_pack_W1(const float* __restrict__ W1, __bf16* __restrict__ Bp) {
    // NT=8, KT=24 : total 8*24*32*16 = 98304
    int tid = blockIdx.x * blockDim.x + threadIdx.x;
    if (tid >= 8 * 24 * 32 * 16) return;
    int e    = tid & 15;
    int lane = (tid >> 4) & 31;
    int kt   = (tid >> 9) % 24;
    int nt   = (tid >> 9) / 24;
    int n = nt * 16 + (lane & 15);
    int k = kt * 32 + (lane >> 4) * 16 + e;
    int head = n >> 5, o = n & 31;                  // col n = head*HID + o
    float v = W1[((size_t)head * IN_DIM + k) * HID + o];
    Bp[tid] = (__bf16)v;
}

__global__ void k_pack_W2(const float* __restrict__ W2, __bf16* __restrict__ Bp) {
    // NT=8, KT=4 : total 8*4*32*16 = 16384
    int tid = blockIdx.x * blockDim.x + threadIdx.x;
    if (tid >= 8 * 4 * 32 * 16) return;
    int e    = tid & 15;
    int lane = (tid >> 4) & 31;
    int kt   = (tid >> 9) % 4;
    int nt   = (tid >> 9) / 4;
    int n = nt * 16 + (lane & 15);
    int k = kt * 32 + (lane >> 4) * 16 + e;
    float v = W2[(size_t)k * OUT_DIM + n];
    Bp[tid] = (__bf16)v;
}

// ---------------------------------------------------------------------------
// WMMA GEMM:  C[M x 128] = A[M x (KT*32)] * B, bf16 in / f32 accumulate.
// 256-thread block = 8 waves.  The whole packed B is cooperatively staged into
// LDS (CDNA5 WGP: 320KB LDS; layer-1 B = 192KB).  Work split per wave:
//   2 M-tiles x 4 N-tiles  (waves 0-3: N cols 0-63 of the block's 8 m-tiles;
//   waves 4-7: N cols 64-127 of the same m-tiles).
// Every B fragment read from LDS feeds two independent WMMAs (2x LDS
// bandwidth amplification - LDS is the binding resource at 256 B/clk/WGP).
// The k-loop is FULLY UNROLLED (KT is constexpr): no loop back-edge means no
// accumulator phi copies - acc stays in one register set, updated in place by
// each WMMA, eliminating the v_mov/v_nop (WMMA->VALU hazard) overhead seen
// with a runtime loop.
// ---------------------------------------------------------------------------
template <int KT>
__global__ __launch_bounds__(256) void k_gemm_bf16(
    const __bf16* __restrict__ A,   // M x K row-major
    const __bf16* __restrict__ Bp,  // packed fragments (see k_pack_*)
    float* __restrict__ C,          // M x 128 row-major
    int mtiles)
{
    constexpr int K = KT * 32;
    constexpr int N = 128;
    constexpr int BELEMS = 8 * KT * 512;         // bf16 elements in packed B

    extern __shared__ __bf16 sB[];

    // ---- cooperative staging: global -> LDS, 16B vectors ----
    {
        constexpr int CHUNKS = BELEMS / (256 * 8);   // 8 bf16 per v8bf
        const int t = threadIdx.x;
#pragma unroll
        for (int i = 0; i < CHUNKS; ++i) {
            int idx = (i * 256 + t) * 8;
            *(v8bf*)(sB + idx) = *(const v8bf*)(Bp + idx);
        }
    }
    __syncthreads();

    const int wave  = threadIdx.x >> 5;
    const int lane  = threadIdx.x & 31;
    const int mpair = wave & 3;                  // which m-tile pair in block
    const int nt0   = (wave >> 2) * 4;           // first of 4 n-tiles for wave
    const int m0    = blockIdx.x * 8 + mpair * 2;
    if (m0 >= mtiles) return;                    // idle tail waves (after barrier)
    const bool has2 = (m0 + 1 < mtiles);

    v8f acc0[4], acc1[4];
#pragma unroll
    for (int i = 0; i < 4; ++i) {
        acc0[i] = (v8f){0.f,0.f,0.f,0.f,0.f,0.f,0.f,0.f};
        acc1[i] = (v8f){0.f,0.f,0.f,0.f,0.f,0.f,0.f,0.f};
    }

    const int rr  = lane & 15;
    const int klo = (lane >> 4) * 16;  // A layout: lanes 0-15 K=0-15, lanes 16-31 K=16-31
    const int row0 = m0 * 16 + rr;
    const int row1 = has2 ? (m0 + 1) * 16 + rr : row0;   // clamp: loads stay in-bounds
    const __bf16* arow0 = A + (size_t)row0 * K + klo;
    const __bf16* arow1 = A + (size_t)row1 * K + klo;
    // lane-local base of this wave's fragment range (n-tiles nt0..nt0+3)
    const __bf16* sBl = sB + (size_t)nt0 * KT * 512 + (size_t)lane * 16;

#pragma unroll
    for (int kt = 0; kt < KT; ++kt) {
        v8bf lo0 = *(const v8bf*)(arow0 + kt * 32);
        v8bf hi0 = *(const v8bf*)(arow0 + kt * 32 + 8);
        v8bf lo1 = *(const v8bf*)(arow1 + kt * 32);
        v8bf hi1 = *(const v8bf*)(arow1 + kt * 32 + 8);
        v16bf a0 = __builtin_shufflevector(lo0, hi0,
            0, 1, 2, 3, 4, 5, 6, 7, 8, 9, 10, 11, 12, 13, 14, 15);
        v16bf a1 = __builtin_shufflevector(lo1, hi1,
            0, 1, 2, 3, 4, 5, 6, 7, 8, 9, 10, 11, 12, 13, 14, 15);

        // each B fragment loaded once from LDS feeds two independent WMMAs
#pragma unroll
        for (int j = 0; j < 4; ++j) {
            v16bf b = *(const v16bf*)(sBl + (size_t)(j * KT + kt) * 512);
            acc0[j] = __builtin_amdgcn_wmma_f32_16x16x32_bf16(
                false, a0, false, b, (short)0, acc0[j], false, false);
            acc1[j] = __builtin_amdgcn_wmma_f32_16x16x32_bf16(
                false, a1, false, b, (short)0, acc1[j], false, false);
        }
    }

    // C/D layout: VGPR r -> M = (lane>=16 ? 8 : 0) + r, N = lane & 15
    const int r0 = (lane >> 4) * 8;
    const int c0 = lane & 15;
#pragma unroll
    for (int j = 0; j < 4; ++j) {
#pragma unroll
        for (int r = 0; r < 8; ++r) {
            C[(size_t)(m0 * 16 + r0 + r) * N + (nt0 + j) * 16 + c0] = acc0[j][r];
        }
    }
    if (has2) {
#pragma unroll
        for (int j = 0; j < 4; ++j) {
#pragma unroll
            for (int r = 0; r < 8; ++r) {
                C[(size_t)((m0 + 1) * 16 + r0 + r) * N + (nt0 + j) * 16 + c0] = acc1[j][r];
            }
        }
    }
}

// ---------------------------------------------------------------------------
// Attention scores:  s_src[n,h] = z[n,h,:] . a[h,:D],  s_dst uses a[h,D:2D]
// ---------------------------------------------------------------------------
__global__ void k_scores(const float* __restrict__ z, const float* __restrict__ a,
                         float* __restrict__ ssrc, float* __restrict__ sdst,
                         int n_nodes, int H, int D) {
    int tid = blockIdx.x * blockDim.x + threadIdx.x;
    if (tid >= n_nodes * H) return;
    int nidx = tid / H, hh = tid % H;
    const float* zz = z + (size_t)nidx * H * D + (size_t)hh * D;
    const float* as = a + (size_t)hh * 2 * D;
    const float* ad = as + D;
    float s0 = 0.f, s1 = 0.f;
    for (int d = 0; d < D; ++d) { float v = zz[d]; s0 += v * as[d]; s1 += v * ad[d]; }
    ssrc[tid] = s0;
    sdst[tid] = s1;
}

// ---------------------------------------------------------------------------
// Edge passes (softmax over incoming edges, then weighted aggregation)
// ---------------------------------------------------------------------------
__device__ inline void atomicMaxFloat(float* addr, float v) {
    if (v >= 0.0f) atomicMax((int*)addr, __float_as_int(v));
    else           atomicMin((unsigned int*)addr, (unsigned int)__float_as_int(v));
}

__device__ inline float lrelu(float x) { return x >= 0.0f ? x : 0.01f * x; }

__global__ void k_edge_max(const int* __restrict__ src, const int* __restrict__ dst,
                           const float* __restrict__ ssrc, const float* __restrict__ sdst,
                           float* __restrict__ emax, int n_edges, int H) {
    int tid = blockIdx.x * blockDim.x + threadIdx.x;
    if (tid >= n_edges * H) return;
    int e = tid / H, hh = tid % H;
    int s = src[e], d = dst[e];
    float x = lrelu(ssrc[s * H + hh] + sdst[d * H + hh]);
    atomicMaxFloat(&emax[d * H + hh], x);
}

__global__ void k_edge_exp(const int* __restrict__ src, const int* __restrict__ dst,
                           const float* __restrict__ ssrc, const float* __restrict__ sdst,
                           const float* __restrict__ emax,
                           float* __restrict__ exbuf, float* __restrict__ den,
                           int n_edges, int H) {
    int tid = blockIdx.x * blockDim.x + threadIdx.x;
    if (tid >= n_edges * H) return;
    int e = tid / H, hh = tid % H;
    int s = src[e], d = dst[e];
    float x = lrelu(ssrc[s * H + hh] + sdst[d * H + hh]);
    float m = emax[d * H + hh];            // finite: this dst has >=1 edge
    float ex = expf(x - m);
    exbuf[tid] = ex;
    atomicAdd(&den[d * H + hh], ex);
}

// 32 lanes per edge, float4 per lane (H*D == 128)
__global__ void k_edge_aggr(const int* __restrict__ src, const int* __restrict__ dst,
                            const float* __restrict__ z, const float* __restrict__ exbuf,
                            const float* __restrict__ den, float* __restrict__ out,
                            int n_edges, int H, int D) {
    int gid  = blockIdx.x * blockDim.x + threadIdx.x;
    int e    = gid >> 5;
    int lane = gid & 31;
    if (e >= n_edges) return;
    int s = src[e], d = dst[e];
    int d0 = lane * 4;                       // all 4 dims in the same head
    int head = d0 / D;
    float alpha = exbuf[(size_t)e * H + head] /
                  fmaxf(den[(size_t)d * H + head], 1e-9f);
    const float4 zv = *(const float4*)(z + (size_t)s * FEAT + d0);
    float* o = out + (size_t)d * FEAT + d0;
    atomicAdd(o + 0, alpha * zv.x);
    atomicAdd(o + 1, alpha * zv.y);
    atomicAdd(o + 2, alpha * zv.z);
    atomicAdd(o + 3, alpha * zv.w);
}

// ---------------------------------------------------------------------------
// Driver
// ---------------------------------------------------------------------------
extern "C" void kernel_launch(void* const* d_in, const int* in_sizes, int n_in,
                              void* d_out, int out_size, void* d_ws, size_t ws_size,
                              hipStream_t stream) {
    const float* h  = (const float*)d_in[0];
    const float* W1 = (const float*)d_in[1];
    const float* a1 = (const float*)d_in[2];
    const float* W2 = (const float*)d_in[3];
    const float* a2 = (const float*)d_in[4];
    const int*   src = (const int*)d_in[5];
    const int*   dst = (const int*)d_in[6];
    float* out = (float*)d_out;

    char* p = (char*)d_ws;
    auto carve = [&](size_t bytes) -> char* {
        char* r = p;
        p += (bytes + 255) & ~(size_t)255;
        return r;
    };

    __bf16* hb   = (__bf16*)carve((size_t)N_NODES * IN_DIM * 2);
    __bf16* W1b  = (__bf16*)carve((size_t)IN_DIM * FEAT * 2);
    float*  z1   = (float*) carve((size_t)N_NODES * FEAT * 4);
    float*  s1s  = (float*) carve((size_t)N_NODES * HEADS * 4);
    float*  s1d  = (float*) carve((size_t)N_NODES * HEADS * 4);
    float*  em1  = (float*) carve((size_t)N_NODES * HEADS * 4);
    float*  dn1  = (float*) carve((size_t)N_NODES * HEADS * 4);
    float*  ex1  = (float*) carve((size_t)N_EDGES * HEADS * 4);
    float*  acc1 = (float*) carve((size_t)N_NODES * FEAT * 4);
    __bf16* h1b  = (__bf16*)carve((size_t)N_NODES * FEAT * 2);
    __bf16* W2b  = (__bf16*)carve((size_t)FEAT * OUT_DIM * 2);
    float*  z2   = (float*) carve((size_t)N_NODES * OUT_DIM * 4);
    float*  s2s  = (float*) carve((size_t)N_NODES * 4);
    float*  s2d  = (float*) carve((size_t)N_NODES * 4);
    float*  em2  = (float*) carve((size_t)N_NODES * 4);
    float*  dn2  = (float*) carve((size_t)N_NODES * 4);
    float*  ex2  = (float*) carve((size_t)N_EDGES * 4);

    const int B = 256;
    const float NEG_INF = -__builtin_inff();
    const int MTILES = N_NODES / 16;               // 3125
    const int GEMM_BLOCKS = (MTILES + 7) / 8;      // 391 (8 m-tiles per block)

    // ---- layer 1 ----
    {
        int n = N_NODES * IN_DIM;
        k_f32_to_bf16<<<(n + B - 1) / B, B, 0, stream>>>(h, hb, n);
    }
    k_pack_W1<<<(8 * 24 * 32 * 16 + B - 1) / B, B, 0, stream>>>(W1, W1b);
    k_gemm_bf16<24><<<GEMM_BLOCKS, 256, 8 * 24 * 512 * 2, stream>>>(hb, W1b, z1, MTILES);

    k_scores<<<(N_NODES * HEADS + B - 1) / B, B, 0, stream>>>(z1, a1, s1s, s1d, N_NODES, HEADS, HID);
    k_fill<<<(N_NODES * HEADS + B - 1) / B, B, 0, stream>>>(em1, NEG_INF, N_NODES * HEADS);
    k_fill<<<(N_NODES * HEADS + B - 1) / B, B, 0, stream>>>(dn1, 0.0f, N_NODES * HEADS);
    k_fill<<<(N_NODES * FEAT + B - 1) / B, B, 0, stream>>>(acc1, 0.0f, N_NODES * FEAT);

    k_edge_max<<<(N_EDGES * HEADS + B - 1) / B, B, 0, stream>>>(src, dst, s1s, s1d, em1, N_EDGES, HEADS);
    k_edge_exp<<<(N_EDGES * HEADS + B - 1) / B, B, 0, stream>>>(src, dst, s1s, s1d, em1, ex1, dn1, N_EDGES, HEADS);
    k_edge_aggr<<<((size_t)N_EDGES * 32 + B - 1) / B, B, 0, stream>>>(src, dst, z1, ex1, dn1, acc1, N_EDGES, HEADS, HID);

    {
        int n = N_NODES * FEAT;
        k_elu_to_bf16<<<(n + B - 1) / B, B, 0, stream>>>(acc1, h1b, n);
    }

    // ---- layer 2 ----
    k_pack_W2<<<(8 * 4 * 32 * 16 + B - 1) / B, B, 0, stream>>>(W2, W2b);
    k_gemm_bf16<4><<<GEMM_BLOCKS, 256, 8 * 4 * 512 * 2, stream>>>(h1b, W2b, z2, MTILES);

    k_scores<<<(N_NODES + B - 1) / B, B, 0, stream>>>(z2, a2, s2s, s2d, N_NODES, 1, OUT_DIM);
    k_fill<<<(N_NODES + B - 1) / B, B, 0, stream>>>(em2, NEG_INF, N_NODES);
    k_fill<<<(N_NODES + B - 1) / B, B, 0, stream>>>(dn2, 0.0f, N_NODES);
    k_fill<<<(N_NODES * OUT_DIM + B - 1) / B, B, 0, stream>>>(out, 0.0f, N_NODES * OUT_DIM);

    k_edge_max<<<(N_EDGES + B - 1) / B, B, 0, stream>>>(src, dst, s2s, s2d, em2, N_EDGES, 1);
    k_edge_exp<<<(N_EDGES + B - 1) / B, B, 0, stream>>>(src, dst, s2s, s2d, em2, ex2, dn2, N_EDGES, 1);
    k_edge_aggr<<<((size_t)N_EDGES * 32 + B - 1) / B, B, 0, stream>>>(src, dst, z2, ex2, dn2, out, N_EDGES, 1, OUT_DIM);
}
